// MultiHeadAttention_78829829750818
// MI455X (gfx1250) — compile-verified, matrix operand order checked by reading, not run on hardware
//
#include <hip/hip_runtime.h>

// ---------------------------------------------------------------------------
// MI455X (gfx1250) multi-head attention, bf16 WMMA pipeline, register-blocked.
// Problem instance fixed at compile time: B=2, S=4096, D=512, H=8, depth=64.
// ---------------------------------------------------------------------------

typedef __attribute__((ext_vector_type(16))) __bf16 v16bf;
typedef __attribute__((ext_vector_type(8)))  __bf16 v8bf;
typedef __attribute__((ext_vector_type(8)))  float  v8f;
typedef __attribute__((ext_vector_type(4)))  float  v4f;

union AB16 { v16bf v; v8bf h[2]; };
union V8F  { v8f v; float f[8]; };
union V4F  { v4f v; float f[4]; };

static constexpr int Bsz = 2;
static constexpr int Seq = 4096;
static constexpr int Dm  = 512;
static constexpr int Hh  = 8;
static constexpr int Mtok = Bsz * Seq;   // 8192

static __device__ __forceinline__ __bf16 f2bf(float x) {
    union { float f; unsigned u; } c; c.f = x;
    unsigned r = (c.u + 0x7FFFu + ((c.u >> 16) & 1u)) >> 16;
    union { unsigned short s; __bf16 b; } o; o.s = (unsigned short)r;
    return o.b;
}

static __device__ __forceinline__ unsigned pack2bf(float a, float b) {
    union { float f; unsigned u; } ca, cb; ca.f = a; cb.f = b;
    unsigned ra = (ca.u + 0x7FFFu + ((ca.u >> 16) & 1u)) >> 16;
    unsigned rb = (cb.u + 0x7FFFu + ((cb.u >> 16) & 1u)) >> 16;
    return (ra & 0xFFFFu) | (rb << 16);
}

#define WMMA_BF16(A, B, C) \
    __builtin_amdgcn_wmma_f32_16x16x32_bf16(false, (A), false, (B), (short)0, (C), false, false)

// ---------------------------------------------------------------------------
// Elementwise fp32 -> bf16 (grid sized exactly; no bounds guard needed)
// ---------------------------------------------------------------------------
__global__ void mha_cvt_bf16(const float* __restrict__ src, __bf16* __restrict__ dst) {
    int i = blockIdx.x * 256 + threadIdx.x;
    dst[i] = f2bf(src[i]);
}

// fp32 [K][N] -> bf16 [N][K], K=N=512 (transpose for contiguous B-fragments)
__global__ void mha_cvt_wT(const float* __restrict__ src, __bf16* __restrict__ dst) {
    int i = blockIdx.x * 256 + threadIdx.x;
    int nn = i >> 9, kk = i & 511;
    dst[i] = f2bf(src[(kk << 9) + nn]);
}

// ---------------------------------------------------------------------------
// Register-blocked WMMA GEMM: out = (A[M,512] @ Bt[512,512]^T + bias) * SCL
// Block tile 128x128 (8 waves, 2x4), wave tile 64x32 (4x2 subtiles).
// MODE 0: bf16 out, head-split  [B,H,S,64]
// MODE 1: bf16 out, head-split T [B,H,64,S]
// MODE 2: fp32 out, plain row-major [M,512]
// ---------------------------------------------------------------------------
template <int MODE>
__global__ __launch_bounds__(256) void mha_gemm_wmma(
    const __bf16* __restrict__ A, const __bf16* __restrict__ Bt,
    const float* __restrict__ bias, void* __restrict__ out, float outScale)
{
    constexpr int K = Dm;
    const int wave = threadIdx.x >> 5;
    const int lane = threadIdx.x & 31;
    const int half = lane >> 4;
    const int mn   = lane & 15;

    constexpr int blocksN = Dm >> 7;            // 4
    const int btm = blockIdx.x >> 2;
    const int btn = blockIdx.x & (blocksN - 1);
    const int waveM = wave >> 2;                // 0..1
    const int waveN = wave & 3;                 // 0..3
    const int rowBase = btm * 128 + waveM * 64;
    const int colBase = btn * 128 + waveN * 32;

    const __bf16* arow0 = A  + (size_t)(rowBase + mn) * K;
    const __bf16* brow0 = Bt + (size_t)(colBase + mn) * K;
    constexpr size_t aStride16 = (size_t)16 * K;

    V8F acc[4][2];
#pragma unroll
    for (int mi = 0; mi < 4; ++mi)
#pragma unroll
        for (int ni = 0; ni < 2; ++ni)
#pragma unroll
            for (int r = 0; r < 8; ++r) acc[mi][ni].f[r] = 0.0f;

    for (int k0 = 0; k0 < K; k0 += 32) {
        AB16 a[4], b[2];
#pragma unroll
        for (int mi = 0; mi < 4; ++mi) {
            const __bf16* p = arow0 + (size_t)mi * aStride16 + k0;
            a[mi].h[0] = *(const v8bf*)(p + half * 8);
            a[mi].h[1] = *(const v8bf*)(p + 16 + half * 8);
        }
#pragma unroll
        for (int ni = 0; ni < 2; ++ni) {
            const __bf16* p = brow0 + (size_t)ni * aStride16 + k0;
            b[ni].h[0] = *(const v8bf*)(p + half * 8);
            b[ni].h[1] = *(const v8bf*)(p + 16 + half * 8);
        }
#pragma unroll
        for (int mi = 0; mi < 4; ++mi)
#pragma unroll
            for (int ni = 0; ni < 2; ++ni)
                acc[mi][ni].v = WMMA_BF16(a[mi].v, b[ni].v, acc[mi][ni].v);
    }

#pragma unroll
    for (int ni = 0; ni < 2; ++ni) {
        const int col = colBase + ni * 16 + mn;
        const float bv = bias[col];
        const int hh = col >> 6, dd = col & 63;
#pragma unroll
        for (int mi = 0; mi < 4; ++mi) {
#pragma unroll
            for (int r = 0; r < 8; ++r) {
                const int row = rowBase + mi * 16 + r + 8 * half;
                const float vv = (acc[mi][ni].f[r] + bv) * outScale;
                if (MODE == 2) {
                    ((float*)out)[((size_t)row << 9) + col] = vv;
                } else {
                    const int bb = row >> 12;          // row / Seq
                    const int ss = row & (Seq - 1);
                    __bf16* o = (__bf16*)out;
                    if (MODE == 0)
                        o[((((size_t)(bb * Hh + hh)) << 12) + ss) * 64 + dd] = f2bf(vv);
                    else
                        o[((((size_t)(bb * Hh + hh)) * 64 + dd) << 12) + ss] = f2bf(vv);
                }
            }
        }
    }
}

// ---------------------------------------------------------------------------
// Flash attention: 32-query tile per wave (2 Q-subtiles), online softmax,
// 32-key steps; K/V fragments reused across both Q-subtiles.
// Qh,Kh: [B,H,S,64] bf16 (Q pre-scaled by 1/sqrt(64)).  Vh: [B,H,64,S] bf16.
// Oattn: [B*S, 512] bf16.  Per-wave private LDS slice -> no barriers;
// EXEC stays all-1s for the whole kernel (row pass uses all 32 lanes).
// ---------------------------------------------------------------------------
__global__ __launch_bounds__(256) void mha_attn_wmma(
    const __bf16* __restrict__ Qh, const __bf16* __restrict__ Kh,
    const __bf16* __restrict__ Vh, __bf16* __restrict__ Oattn)
{
    __shared__ __align__(16) float    sS[8][32][32];   // fp32 scores (pre-scaled)
    __shared__ __align__(16) unsigned sP[8][32][16];   // exp()'d P, packed 2xbf16
    __shared__ __align__(16) float    sC[8][32];       // per-row broadcast

    const int wave = threadIdx.x >> 5;
    const int lane = threadIdx.x & 31;
    const int half = lane >> 4;
    const int mn   = lane & 15;

    constexpr int qTiles = Seq >> 5;            // 128
    const int tile = blockIdx.x * 8 + wave;
    const int bh = tile >> 7;                   // tile / qTiles
    const int qb = (tile & (qTiles - 1)) << 5;

    const __bf16* Qb = Qh + (size_t)bh * Seq * 64;
    const __bf16* Kb = Kh + (size_t)bh * Seq * 64;
    const __bf16* Vb = Vh + (size_t)bh * 64 * Seq;

    // Q fragments: 2 query subtiles x 2 depth chunks
    AB16 aq[2][2];
#pragma unroll
    for (int qt = 0; qt < 2; ++qt) {
        const __bf16* qrow = Qb + (size_t)(qb + qt * 16 + mn) * 64;
#pragma unroll
        for (int kc = 0; kc < 2; ++kc) {
            aq[qt][kc].h[0] = *(const v8bf*)(qrow + kc * 32 + half * 8);
            aq[qt][kc].h[1] = *(const v8bf*)(qrow + kc * 32 + 16 + half * 8);
        }
    }

    V8F acc[2][4];
#pragma unroll
    for (int qt = 0; qt < 2; ++qt)
#pragma unroll
        for (int dt = 0; dt < 4; ++dt)
#pragma unroll
            for (int r = 0; r < 8; ++r) acc[qt][dt].f[r] = 0.0f;

    float m_i = -3.0e38f;   // per-lane row state: row = lane (0..31)
    float l_i = 0.0f;

    for (int kt = 0; kt < Seq; kt += 32) {
        // ---- scores: S(32x32) = Q(32x64) @ K_tile^T(64x32) ----
        V8F s[2][2];
#pragma unroll
        for (int qt = 0; qt < 2; ++qt)
#pragma unroll
            for (int j = 0; j < 2; ++j)
#pragma unroll
                for (int r = 0; r < 8; ++r) s[qt][j].f[r] = 0.0f;

#pragma unroll
        for (int j = 0; j < 2; ++j) {           // key halves (16 keys each)
            AB16 bk0, bk1;
            const __bf16* krow = Kb + (size_t)(kt + j * 16 + mn) * 64;
            bk0.h[0] = *(const v8bf*)(krow + half * 8);
            bk0.h[1] = *(const v8bf*)(krow + 16 + half * 8);
            bk1.h[0] = *(const v8bf*)(krow + 32 + half * 8);
            bk1.h[1] = *(const v8bf*)(krow + 48 + half * 8);
#pragma unroll
            for (int qt = 0; qt < 2; ++qt) {
                s[qt][j].v = WMMA_BF16(aq[qt][0].v, bk0.v, s[qt][j].v);
                s[qt][j].v = WMMA_BF16(aq[qt][1].v, bk1.v, s[qt][j].v);
            }
        }

        // ---- D-fragment -> LDS (layout transpose for P) ----
#pragma unroll
        for (int qt = 0; qt < 2; ++qt)
#pragma unroll
            for (int j = 0; j < 2; ++j)
#pragma unroll
                for (int r = 0; r < 8; ++r)
                    sS[wave][qt * 16 + r + 8 * half][j * 16 + mn] = s[qt][j].f[r];

        // ---- online softmax row pass: all 32 lanes, row = lane ----
        {
            const int row = lane;
            V4F t[8];
#pragma unroll
            for (int j = 0; j < 8; ++j)
                t[j].v = *(const v4f*)(&sS[wave][row][j * 4]);

            float tmax = -3.0e38f;
#pragma unroll
            for (int j = 0; j < 8; ++j)
#pragma unroll
                for (int e = 0; e < 4; ++e) tmax = fmaxf(tmax, t[j].f[e]);

            const float mnew = fmaxf(m_i, tmax);
            const float corr = __expf(m_i - mnew);
            float sum = 0.0f;
#pragma unroll
            for (int j = 0; j < 8; ++j) {
#pragma unroll
                for (int e = 0; e < 4; e += 2) {
                    const float e0 = __expf(t[j].f[e]     - mnew);
                    const float e1 = __expf(t[j].f[e + 1] - mnew);
                    sum += e0 + e1;
                    sP[wave][row][j * 2 + (e >> 1)] = pack2bf(e0, e1);
                }
            }
            l_i = l_i * corr + sum;
            m_i = mnew;
            sC[wave][row] = corr;
        }

        // ---- rescale accumulators by per-row correction ----
#pragma unroll
        for (int qt = 0; qt < 2; ++qt)
#pragma unroll
            for (int r = 0; r < 8; ++r) {
                const float c = sC[wave][qt * 16 + r + 8 * half];
#pragma unroll
                for (int dt = 0; dt < 4; ++dt) acc[qt][dt].f[r] *= c;
            }

        // ---- P A-fragments from LDS ----
        AB16 ap[2];
#pragma unroll
        for (int qt = 0; qt < 2; ++qt) {
            const __bf16* prow = (const __bf16*)&sP[wave][qt * 16 + mn][0];
            ap[qt].h[0] = *(const v8bf*)(prow + half * 8);
            ap[qt].h[1] = *(const v8bf*)(prow + 16 + half * 8);
        }

        // ---- out += P(32x32) @ V_tile(32x64); V frags shared by both Q tiles ----
#pragma unroll
        for (int dt = 0; dt < 4; ++dt) {
            AB16 bv;
            const __bf16* vrow = Vb + (size_t)((dt * 16 + mn) << 12) + kt;
            bv.h[0] = *(const v8bf*)(vrow + half * 8);
            bv.h[1] = *(const v8bf*)(vrow + 16 + half * 8);
#pragma unroll
            for (int qt = 0; qt < 2; ++qt)
                acc[qt][dt].v = WMMA_BF16(ap[qt].v, bv.v, acc[qt][dt].v);
        }
    }

    // ---- finalize: divide by row sums, write bf16 [B*S, 512] ----
    sC[wave][lane] = 1.0f / l_i;

    const int b = bh >> 3, h = bh & 7;
#pragma unroll
    for (int qt = 0; qt < 2; ++qt)
#pragma unroll
        for (int r = 0; r < 8; ++r) {
            const int row = qt * 16 + r + 8 * half;
            const float inv = sC[wave][row];
            const size_t tok = ((size_t)b << 12) + qb + row;
#pragma unroll
            for (int dt = 0; dt < 4; ++dt)
                Oattn[(tok << 9) + h * 64 + dt * 16 + mn] = f2bf(acc[qt][dt].f[r] * inv);
        }
}

// ---------------------------------------------------------------------------
// Launch
// ---------------------------------------------------------------------------
extern "C" void kernel_launch(void* const* d_in, const int* in_sizes, int n_in,
                              void* d_out, int out_size, void* d_ws, size_t ws_size,
                              hipStream_t stream) {
    const float* q    = (const float*)d_in[0];
    const float* k    = (const float*)d_in[1];
    const float* v    = (const float*)d_in[2];
    const float* wq_w = (const float*)d_in[3];
    const float* wq_b = (const float*)d_in[4];
    const float* wk_w = (const float*)d_in[5];
    const float* wk_b = (const float*)d_in[6];
    const float* wv_w = (const float*)d_in[7];
    const float* wv_b = (const float*)d_in[8];
    const float* wo_w = (const float*)d_in[9];
    const float* wo_b = (const float*)d_in[10];

    const size_t actB = (size_t)Mtok * Dm * 2;   // 8 MB bf16 activation
    const size_t wB   = (size_t)Dm * Dm * 2;     // 512 KB bf16 weight

    char* ws = (char*)d_ws;
    __bf16* xq   = (__bf16*)(ws);
    __bf16* xk   = (__bf16*)(ws + actB);
    __bf16* xv   = (__bf16*)(ws + 2 * actB);
    __bf16* wqT  = (__bf16*)(ws + 3 * actB);
    __bf16* wkT  = (__bf16*)(ws + 3 * actB + wB);
    __bf16* wvT  = (__bf16*)(ws + 3 * actB + 2 * wB);
    __bf16* woT  = (__bf16*)(ws + 3 * actB + 3 * wB);
    __bf16* qh   = (__bf16*)(ws + 3 * actB + 4 * wB);
    __bf16* kh   = (__bf16*)(ws + 4 * actB + 4 * wB);
    __bf16* vh   = (__bf16*)(ws + 5 * actB + 4 * wB);
    __bf16* attn = (__bf16*)(ws + 6 * actB + 4 * wB);

    const int nAct = Mtok * Dm;                  // 4194304, /256 exact
    mha_cvt_bf16<<<nAct / 256, 256, 0, stream>>>(q, xq);
    mha_cvt_bf16<<<nAct / 256, 256, 0, stream>>>(k, xk);
    mha_cvt_bf16<<<nAct / 256, 256, 0, stream>>>(v, xv);

    const int nW = Dm * Dm;                      // 262144, /256 exact
    mha_cvt_wT<<<nW / 256, 256, 0, stream>>>(wq_w, wqT);
    mha_cvt_wT<<<nW / 256, 256, 0, stream>>>(wk_w, wkT);
    mha_cvt_wT<<<nW / 256, 256, 0, stream>>>(wv_w, wvT);
    mha_cvt_wT<<<nW / 256, 256, 0, stream>>>(wo_w, woT);

    const float qScale = 0.125f;                 // 1/sqrt(depth), folded into Q proj
    const int gemmBlocks = (Mtok / 128) * (Dm / 128);   // 256
    mha_gemm_wmma<0><<<gemmBlocks, 256, 0, stream>>>(xq, wqT, wq_b, qh, qScale);
    mha_gemm_wmma<0><<<gemmBlocks, 256, 0, stream>>>(xk, wkT, wk_b, kh, 1.0f);
    mha_gemm_wmma<1><<<gemmBlocks, 256, 0, stream>>>(xv, wvT, wv_b, vh, 1.0f);

    const int attnBlocks = (Bsz * Hh * (Seq / 32)) / 8; // 256
    mha_attn_wmma<<<attnBlocks, 256, 0, stream>>>(qh, kh, vh, attn);

    mha_gemm_wmma<2><<<gemmBlocks, 256, 0, stream>>>(attn, woT, wo_b, d_out, 1.0f);
}
